// SpatialAwareMultiHeadedAttention_197568496209
// MI455X (gfx1250) — matrix operand order repaired and checked by reading, not compiled, not run on hardware
//
#include <hip/hip_runtime.h>
#include <hip/hip_bf16.h>
#include <cstdint>
#include <cstddef>

typedef __bf16 bf16;
typedef __attribute__((ext_vector_type(16))) __bf16 v16bf;
typedef __attribute__((ext_vector_type(2)))  __bf16 bf16x2;
typedef __attribute__((ext_vector_type(8)))  float  v8f;

constexpr int B_ = 8, L_ = 1024, D_ = 512, H_ = 8, DK_ = 64;
constexpr int KDIM = D_;          // GEMM reduction dim for projections

// A-fragment (16x32 bf16) pair offset inside a row for vgpr-pair j, lane-half g.
// Layout (ISA 7.12.2): lanes 0-15: V0..3 -> K 0..7, V4..7 -> K 16..23;
//                      lanes 16-31: V0..3 -> K 8..15, V4..7 -> K 24..31.
__device__ __forceinline__ int apair(int j, int g) {
  return (j < 4) ? (8 * g + 2 * j) : (8 + 8 * g + 2 * j);
}

__device__ __forceinline__ float rmax16(float v) {
#pragma unroll
  for (int m = 1; m < 16; m <<= 1) v = fmaxf(v, __shfl_xor(v, m, 32));
  return v;
}
__device__ __forceinline__ float rsum16(float v) {
#pragma unroll
  for (int m = 1; m < 16; m <<= 1) v += __shfl_xor(v, m, 32);
  return v;
}

// ---- CDNA5 async global->LDS (ASYNCcnt-tracked DMA into LDS) ------------
// Generic pointers to __shared__ carry the LDS address in their low 32 bits
// (ISA 10.2: LDS aperture uses addr[31:0]).
__device__ __forceinline__ unsigned lds_addr(const void* p) {
  return (unsigned)(uintptr_t)p;
}
__device__ __forceinline__ void async_g2l_b128(unsigned loff, const bf16* g) {
  asm volatile("global_load_async_to_lds_b128 %0, %1, off"
               :: "v"(loff), "v"((uintptr_t)g)
               : "memory");
}
__device__ __forceinline__ void wait_async0() {
  asm volatile("s_wait_asynccnt 0x0" ::: "memory");
}

// ---------------------------------------------------------------------------
// f32 -> bf16 conversion (grid-stride)
// ---------------------------------------------------------------------------
__global__ void cvt_f32_bf16(const float* __restrict__ in, bf16* __restrict__ out, int n) {
  for (int i = blockIdx.x * blockDim.x + threadIdx.x; i < n; i += gridDim.x * blockDim.x)
    out[i] = (bf16)in[i];
}

// ---------------------------------------------------------------------------
// Projection GEMM: Y = X(M,K) * W(N,K)^T + bias, one 16x16 tile per wave.
// MODE 0: bf16 head-split (B,H,L,DK)            (Q, K projections)
// MODE 1: bf16 head-split transposed (B,H,DK,L) (V projection)
// MODE 2: f32 (B,L,D)                           (output projection)
// ---------------------------------------------------------------------------
template <int MODE>
__global__ void __launch_bounds__(256)
proj_wmma(const bf16* __restrict__ X, const bf16* __restrict__ W,
          const float* __restrict__ bias, void* __restrict__ outp) {
  const int lane = threadIdx.x & 31;
  const int g = lane >> 4, ln = lane & 15;
  const int wave = blockIdx.x * (blockDim.x >> 5) + (threadIdx.x >> 5);
  const int ntiles = D_ / 16;                 // 32
  const int mtile = wave / ntiles;
  const int ntile = wave % ntiles;
  const int rbase = mtile * 16;
  const int cbase = ntile * 16;

  const bf16* arow = X + (size_t)(rbase + ln) * KDIM;   // A: lane = row
  const bf16* brow = W + (size_t)(cbase + ln) * KDIM;   // B: lane = out column, W[n][k]

  v8f acc = {};
#pragma unroll 4
  for (int ks = 0; ks < KDIM; ks += 32) {
    v16bf a, b;
#pragma unroll
    for (int j = 0; j < 8; ++j) {
      bf16x2 pa = *reinterpret_cast<const bf16x2*>(arow + ks + apair(j, g));
      bf16x2 pb = *reinterpret_cast<const bf16x2*>(brow + ks + apair(j, g));
      a[2 * j] = pa[0]; a[2 * j + 1] = pa[1];
      b[2 * j] = pb[0]; b[2 * j + 1] = pb[1];
    }
    acc = __builtin_amdgcn_wmma_f32_16x16x32_bf16(false, a, false, b,
                                                  (short)0, acc, false, false);
  }

  const float bc = bias[cbase + ln];
#pragma unroll
  for (int j = 0; j < 8; ++j) {
    const int r = rbase + 8 * g + j;     // global output row (b*L + pos)
    const int c = cbase + ln;            // global output col (h*DK + dk)
    const float v = acc[j] + bc;
    if (MODE == 2) {
      ((float*)outp)[(size_t)r * D_ + c] = v;
    } else {
      const int bb = r >> 10, pos = r & (L_ - 1);
      const int hh = c >> 6, dk = c & (DK_ - 1);
      if (MODE == 0)
        ((bf16*)outp)[(((size_t)(bb * H_ + hh)) * L_ + pos) * DK_ + dk] = (bf16)v;
      else
        ((bf16*)outp)[(((size_t)(bb * H_ + hh)) * DK_ + dk) * L_ + pos] = (bf16)v;
    }
  }
}

// ---------------------------------------------------------------------------
// Fused attention. One block (8 waves) works on one (b,h) and 8 adjacent
// 16-query tiles; K/Vt chunks are staged through double-buffered LDS with
// CDNA5 async global->LDS DMA so the K/V streams are fetched once per block
// instead of once per wave.
// ---------------------------------------------------------------------------
__global__ void __launch_bounds__(256)
attn_wmma(const bf16* __restrict__ Qh, const bf16* __restrict__ Kh,
          const bf16* __restrict__ Vt, const float* __restrict__ dist,
          const unsigned char* __restrict__ mask, const float* __restrict__ logwb,
          bf16* __restrict__ Xo) {
  __shared__ __align__(16) bf16 Kbuf[2][32][64];   // (key, feat)   4KB per buf
  __shared__ __align__(16) bf16 Vbuf[2][64][32];   // (dk, key)     4KB per buf
  __shared__ __align__(16) bf16 Plds[8][16][32];   // per-wave P tile

  const int tid = threadIdx.x;
  const int lane = tid & 31;
  const int g = lane >> 4, ln = lane & 15;
  const int wslot = tid >> 5;

  const int bh = blockIdx.x >> 3;             // 0..63  (b*H + h)
  const int qgroup = blockIdx.x & 7;
  const int qbase = (qgroup * 8 + wslot) * 16;
  const int b = bh >> 3, h = bh & 7;

  const bf16* Qp = Qh + ((size_t)bh * L_ + qbase) * DK_;
  const bf16* Kp = Kh + (size_t)bh * L_ * DK_;
  const bf16* Vp = Vt + (size_t)bh * DK_ * L_;
  const float* Dp = dist + (size_t)b * L_ * L_;
  const unsigned char* Mp = mask + (size_t)b * L_ * L_;
  const float wb = __expf(logwb[h]);

  // Per-thread async staging: 16B granule of K chunk + 16B granule of V chunk.
  // K chunk (32x64 bf16): granule t -> row t>>3, cols (t&7)*8;   LDS off = 16*t
  // V chunk (64x32 bf16): granule t -> row t>>2, cols (t&3)*8;   LDS off = 16*t
  auto prefetch = [&](int kc, int nb) {
    async_g2l_b128(lds_addr(&Kbuf[nb][0][0]) + tid * 16,
                   Kp + (size_t)(kc + (tid >> 3)) * DK_ + (tid & 7) * 8);
    async_g2l_b128(lds_addr(&Vbuf[nb][0][0]) + tid * 16,
                   Vp + (size_t)(tid >> 2) * L_ + kc + (tid & 3) * 8);
  };

  // Q fragments: two k-steps of 32 features, loaded once, reused for all chunks
  v16bf aq[2];
#pragma unroll
  for (int s = 0; s < 2; ++s)
#pragma unroll
    for (int j = 0; j < 8; ++j) {
      bf16x2 p = *reinterpret_cast<const bf16x2*>(Qp + (size_t)ln * DK_ + 32 * s + apair(j, g));
      aq[s][2 * j] = p[0]; aq[s][2 * j + 1] = p[1];
    }

  float mrun[8], lrun[8];
  v8f o[4] = {};
#pragma unroll
  for (int j = 0; j < 8; ++j) { mrun[j] = -1e30f; lrun[j] = 0.f; }

  prefetch(0, 0);

  for (int kc = 0, it = 0; kc < L_; kc += 32, ++it) {
    const int cur = it & 1;
    wait_async0();        // our chunk-`it` DMA granules are in LDS
    __syncthreads();      // everyone's granules visible; prev readers of buf^1 done
    if (kc + 32 < L_) prefetch(kc + 32, cur ^ 1);

    // ---- scores: two 16x16 tiles over this 32-key chunk (K from LDS) ----
    v8f st[2];
#pragma unroll
    for (int t = 0; t < 2; ++t) {
      const bf16* krow = &Kbuf[cur][16 * t + ln][0];
      v16bf b0, b1;
#pragma unroll
      for (int j = 0; j < 8; ++j) {
        bf16x2 p0 = *reinterpret_cast<const bf16x2*>(krow + apair(j, g));
        bf16x2 p1 = *reinterpret_cast<const bf16x2*>(krow + 32 + apair(j, g));
        b0[2 * j] = p0[0]; b0[2 * j + 1] = p0[1];
        b1[2 * j] = p1[0]; b1[2 * j + 1] = p1[1];
      }
      v8f s = {};
      s = __builtin_amdgcn_wmma_f32_16x16x32_bf16(false, aq[0], false, b0, (short)0, s, false, false);
      s = __builtin_amdgcn_wmma_f32_16x16x32_bf16(false, aq[1], false, b1, (short)0, s, false, false);
      st[t] = s;
    }

    // ---- distance bias + mask + online softmax update ----
#pragma unroll
    for (int j = 0; j < 8; ++j) {
      const int q = qbase + 8 * g + j;
      const size_t r0 = (size_t)q * L_ + kc + ln;
      if (j == 0 && kc + 32 < L_) __builtin_prefetch(&Dp[r0 + 32], 0, 0);
      float s0 = st[0][j] * 0.125f - Dp[r0] * wb;
      float s1 = st[1][j] * 0.125f - Dp[r0 + 16] * wb;
      if (Mp[r0]) s0 = -1e9f;
      if (Mp[r0 + 16]) s1 = -1e9f;
      const float rm = rmax16(fmaxf(s0, s1));
      const float mn = fmaxf(mrun[j], rm);
      const float sc = __expf(mrun[j] - mn);
      const float p0 = __expf(s0 - mn);
      const float p1 = __expf(s1 - mn);
      lrun[j] = lrun[j] * sc + rsum16(p0 + p1);
      mrun[j] = mn;
#pragma unroll
      for (int t4 = 0; t4 < 4; ++t4) o[t4][j] *= sc;
      Plds[wslot][8 * g + j][ln] = (bf16)p0;          // cols 0..15 of chunk
      Plds[wslot][8 * g + j][16 + ln] = (bf16)p1;     // cols 16..31
    }

    // ---- reload P as A-fragment (lane = query row) ----
    v16bf ap;
    const bf16* prow = &Plds[wslot][ln][0];
#pragma unroll
    for (int j = 0; j < 8; ++j) {
      bf16x2 p = *reinterpret_cast<const bf16x2*>(prow + apair(j, g));
      ap[2 * j] = p[0]; ap[2 * j + 1] = p[1];
    }

    // ---- O += P * V  (V from LDS; lane column = dk, k = key pos) ----
#pragma unroll
    for (int t4 = 0; t4 < 4; ++t4) {
      const bf16* vrow = &Vbuf[cur][16 * t4 + ln][0];
      v16bf bv;
#pragma unroll
      for (int j = 0; j < 8; ++j) {
        bf16x2 p = *reinterpret_cast<const bf16x2*>(vrow + apair(j, g));
        bv[2 * j] = p[0]; bv[2 * j + 1] = p[1];
      }
      o[t4] = __builtin_amdgcn_wmma_f32_16x16x32_bf16(false, ap, false, bv,
                                                      (short)0, o[t4], false, false);
    }
  }

  // ---- normalize rows and store attention output bf16 (B,L,D) head-merged ----
#pragma unroll
  for (int j = 0; j < 8; ++j) {
    const int q = qbase + 8 * g + j;
    const float inv = 1.0f / lrun[j];
#pragma unroll
    for (int t4 = 0; t4 < 4; ++t4) {
      const float v = o[t4][j] * inv;
      Xo[((size_t)(b * L_) + q) * D_ + h * DK_ + 16 * t4 + ln] = (bf16)v;
    }
  }
}

// ---------------------------------------------------------------------------
extern "C" void kernel_launch(void* const* d_in, const int* in_sizes, int n_in,
                              void* d_out, int out_size, void* d_ws, size_t ws_size,
                              hipStream_t stream) {
  const float* q    = (const float*)d_in[0];
  const float* k    = (const float*)d_in[1];
  const float* v    = (const float*)d_in[2];
  const float* dist = (const float*)d_in[3];
  const unsigned char* mask = (const unsigned char*)d_in[4];
  const float* Wq = (const float*)d_in[5];
  const float* bq = (const float*)d_in[6];
  const float* Wk = (const float*)d_in[7];
  const float* bk = (const float*)d_in[8];
  const float* Wv = (const float*)d_in[9];
  const float* bv = (const float*)d_in[10];
  const float* Wo = (const float*)d_in[11];
  const float* bo = (const float*)d_in[12];
  const float* lw = (const float*)d_in[13];

  const size_t SZ_X = (size_t)B_ * L_ * D_;   // 4,194,304
  const size_t SZ_W = (size_t)D_ * D_;        //   262,144

  bf16* ws  = (bf16*)d_ws;
  bf16* qb  = ws;
  bf16* kb  = qb + SZ_X;
  bf16* vbb = kb + SZ_X;
  bf16* wqb = vbb + SZ_X;
  bf16* wkb = wqb + SZ_W;
  bf16* wvb = wkb + SZ_W;
  bf16* wob = wvb + SZ_W;
  bf16* Qh  = wob + SZ_W;
  bf16* Kh  = Qh + SZ_X;
  bf16* Vtb = Kh + SZ_X;
  bf16* Xo  = Vtb + SZ_X;

  // 1) bf16 conversions
  cvt_f32_bf16<<<dim3(1024), dim3(256), 0, stream>>>(q, qb, (int)SZ_X);
  cvt_f32_bf16<<<dim3(1024), dim3(256), 0, stream>>>(k, kb, (int)SZ_X);
  cvt_f32_bf16<<<dim3(1024), dim3(256), 0, stream>>>(v, vbb, (int)SZ_X);
  cvt_f32_bf16<<<dim3(256), dim3(256), 0, stream>>>(Wq, wqb, (int)SZ_W);
  cvt_f32_bf16<<<dim3(256), dim3(256), 0, stream>>>(Wk, wkb, (int)SZ_W);
  cvt_f32_bf16<<<dim3(256), dim3(256), 0, stream>>>(Wv, wvb, (int)SZ_W);
  cvt_f32_bf16<<<dim3(256), dim3(256), 0, stream>>>(Wo, wob, (int)SZ_W);

  // 2) projections: (8192/16)*(512/16) = 16384 waves -> 2048 blocks of 8 waves
  const dim3 pg(2048), pb(256);
  proj_wmma<0><<<pg, pb, 0, stream>>>(qb, wqb, bq, (void*)Qh);
  proj_wmma<0><<<pg, pb, 0, stream>>>(kb, wkb, bk, (void*)Kh);
  proj_wmma<1><<<pg, pb, 0, stream>>>(vbb, wvb, bv, (void*)Vtb);

  // 3) fused attention: 64 (b,h) x 8 q-groups = 512 blocks of 8 waves
  attn_wmma<<<dim3(512), dim3(256), 0, stream>>>(Qh, Kh, Vtb, dist, mask, lw, Xo);

  // 4) output projection, f32 epilogue into d_out
  proj_wmma<2><<<pg, pb, 0, stream>>>(Xo, wob, bo, d_out);
}
